// GAT_18184891531290
// MI455X (gfx1250) — compile-verified, matrix operand order checked by reading, not run on hardware
//
#include <hip/hip_runtime.h>
#include <math.h>

// GAT layer for MI455X (gfx1250, wave32, WMMA).
//
// Pipeline: init -> qkv GEMM (WMMA f16->f32, 16x64 per wave) ->
//           wave-per-edge scores + segment max -> exp + segment sum ->
//           wave-per-edge weighted scatter-add -> out GEMM (WMMA) with
//           bias/ReLU/residual epilogue.

typedef __attribute__((ext_vector_type(16))) _Float16 v16h;
typedef __attribute__((ext_vector_type(8)))  float    v8f;

#define HEADS 4
#define DIM   128

// ---------------------------------------------------------------------------
// WMMA tile loaders (wave32 layouts per CDNA5 ISA 7.12.2)
// A: 16x32 f16. lane l: m = l&15, half = l>>4; element i holds
//    K = kk + (i<8 ? half*8+i : 16 + half*8 + (i-8))  -> two contiguous 8-float runs
// B: 32x16 f16. lane l: n = l&15, half = l>>4; same K mapping per element.
// C/D: 16x16 f32. lane l: n = l&15; vgpr r holds m = (l>>4)*8 + r.
// ---------------------------------------------------------------------------
__device__ __forceinline__ v16h load_a_tile(const float* __restrict__ xrow, int kk, int lh) {
  const float* p0 = xrow + kk + lh * 8;
  const float* p1 = xrow + kk + 16 + lh * 8;
  float4 a0 = *(const float4*)(p0);
  float4 a1 = *(const float4*)(p0 + 4);
  float4 a2 = *(const float4*)(p1);
  float4 a3 = *(const float4*)(p1 + 4);
  v16h a;
  a[0]  = (_Float16)a0.x; a[1]  = (_Float16)a0.y; a[2]  = (_Float16)a0.z; a[3]  = (_Float16)a0.w;
  a[4]  = (_Float16)a1.x; a[5]  = (_Float16)a1.y; a[6]  = (_Float16)a1.z; a[7]  = (_Float16)a1.w;
  a[8]  = (_Float16)a2.x; a[9]  = (_Float16)a2.y; a[10] = (_Float16)a2.z; a[11] = (_Float16)a2.w;
  a[12] = (_Float16)a3.x; a[13] = (_Float16)a3.y; a[14] = (_Float16)a3.z; a[15] = (_Float16)a3.w;
  return a;
}

__device__ __forceinline__ v16h load_b_tile(const float* __restrict__ W, int kk, int lh,
                                            int ncol, int ldw) {
  v16h b;
#pragma unroll
  for (int i = 0; i < 8; ++i) {
    b[i]     = (_Float16)W[(size_t)(kk + lh * 8 + i) * ldw + ncol];
    b[i + 8] = (_Float16)W[(size_t)(kk + 16 + lh * 8 + i) * ldw + ncol];
  }
  return b;
}

// ---------------------------------------------------------------------------
// Workspace init: agg = 0, denom = 0, m = -FLT_MAX
// ---------------------------------------------------------------------------
__global__ void init_ws(float* __restrict__ m, float* __restrict__ denom,
                        float* __restrict__ agg, int n) {
  int i = blockIdx.x * blockDim.x + threadIdx.x;
  if (i < n * DIM) agg[i] = 0.0f;
  if (i < n * HEADS) {
    m[i]     = -3.402823466e38f;
    denom[i] = 0.0f;
  }
}

// ---------------------------------------------------------------------------
// q/k/v projection: one wave per 16(M)x64(N) output block -> A chunk loaded
// once per K step, 4 WMMAs against it. 6 column groups (3 weights x 2).
// ---------------------------------------------------------------------------
__global__ void qkv_wmma(const float* __restrict__ x,
                         const float* __restrict__ Wt,
                         const float* __restrict__ Wsm,
                         const float* __restrict__ Wcm,
                         float* __restrict__ q, float* __restrict__ kmat,
                         float* __restrict__ vmat, int n) {
  int gwave = (blockIdx.x * blockDim.x + threadIdx.x) >> 5;
  int lane  = threadIdx.x & 31;
  int tileM = gwave / 6;
  int g     = gwave % 6;      // column group: weight = g>>1, half = g&1
  int row0  = tileM * 16;
  if (row0 >= n) return;  // wave-uniform

  int wsel = g >> 1;
  const float* W    = (wsel == 0) ? Wt : ((wsel == 1) ? Wsm : Wcm);
  float*       outp = (wsel == 0) ? q  : ((wsel == 1) ? kmat : vmat);
  int colbase = (g & 1) * 64;

  int lm = lane & 15;
  int lh = lane >> 4;
  int arow = row0 + lm;
  if (arow >= n) arow = n - 1;  // clamp (n is a multiple of 16 here anyway)
  const float* xrow = x + (size_t)arow * DIM;

  v8f acc0 = {}, acc1 = {}, acc2 = {}, acc3 = {};
#pragma unroll
  for (int kk = 0; kk < DIM; kk += 32) {
    v16h a  = load_a_tile(xrow, kk, lh);
    v16h b0 = load_b_tile(W, kk, lh, colbase + 0  + lm, DIM);
    v16h b1 = load_b_tile(W, kk, lh, colbase + 16 + lm, DIM);
    v16h b2 = load_b_tile(W, kk, lh, colbase + 32 + lm, DIM);
    v16h b3 = load_b_tile(W, kk, lh, colbase + 48 + lm, DIM);
    acc0 = __builtin_amdgcn_wmma_f32_16x16x32_f16(false, a, false, b0, (short)0, acc0, false, false);
    acc1 = __builtin_amdgcn_wmma_f32_16x16x32_f16(false, a, false, b1, (short)0, acc1, false, false);
    acc2 = __builtin_amdgcn_wmma_f32_16x16x32_f16(false, a, false, b2, (short)0, acc2, false, false);
    acc3 = __builtin_amdgcn_wmma_f32_16x16x32_f16(false, a, false, b3, (short)0, acc3, false, false);
  }

#pragma unroll
  for (int r = 0; r < 8; ++r) {
    int orow = row0 + lh * 8 + r;
    if (orow < n) {
      float* orowp = outp + (size_t)orow * DIM + colbase + lm;
      orowp[0]  = acc0[r];
      orowp[16] = acc1[r];
      orowp[32] = acc2[r];
      orowp[48] = acc3[r];
    }
  }
}

// ---------------------------------------------------------------------------
// Per-edge attention scores + segment max. One wave per edge: lanes read the
// full 128-dim q[dst] / k[src] rows coalesced (float4/lane), head-wise dot
// finished with 3 shfl_xor steps inside 8-lane groups (head = lane>>3).
// ---------------------------------------------------------------------------
__global__ void edge_scores(const float* __restrict__ q, const float* __restrict__ kmat,
                            const int* __restrict__ src, const int* __restrict__ dst,
                            float* __restrict__ sc, float* __restrict__ m, int E) {
  int gwave = (blockIdx.x * blockDim.x + threadIdx.x) >> 5;
  int lane  = threadIdx.x & 31;
  if (gwave >= E) return;  // wave-uniform
  int s = src[gwave];
  int d = dst[gwave];
  float4 qv = ((const float4*)(q    + (size_t)d * DIM))[lane];
  float4 kv = ((const float4*)(kmat + (size_t)s * DIM))[lane];
  float p = qv.x * kv.x + qv.y * kv.y + qv.z * kv.z + qv.w * kv.w;
  p += __shfl_xor(p, 1);
  p += __shfl_xor(p, 2);
  p += __shfl_xor(p, 4);
  if ((lane & 7) == 0) {
    int h = lane >> 3;
    sc[(size_t)gwave * HEADS + h] = p;  // TAU == 1
    atomicMax(&m[(size_t)d * HEADS + h], p);
  }
}

// ---------------------------------------------------------------------------
// exp(score - segmax) + segment sum of exps (thread per edge-head)
// ---------------------------------------------------------------------------
__global__ void edge_exp(const int* __restrict__ dst, float* __restrict__ sc,
                         const float* __restrict__ m, float* __restrict__ denom, int E) {
  int t = blockIdx.x * blockDim.x + threadIdx.x;
  if (t >= E * HEADS) return;
  int e = t >> 2;
  int h = t & 3;
  int d = dst[e];
  float ex = __expf(sc[t] - m[(size_t)d * HEADS + h]);
  sc[t] = ex;
  atomicAdd(&denom[(size_t)d * HEADS + h], ex);
}

// ---------------------------------------------------------------------------
// alpha-weighted scatter-add of v[src] into agg[dst]. One wave per edge:
// coalesced float4/lane read of the v row, 4 fp32 atomics per lane.
// ---------------------------------------------------------------------------
__global__ void edge_scatter(const int* __restrict__ src, const int* __restrict__ dst,
                             const float* __restrict__ sc, const float* __restrict__ denom,
                             const float* __restrict__ vmat, float* __restrict__ agg, int E) {
  int gwave = (blockIdx.x * blockDim.x + threadIdx.x) >> 5;
  int lane  = threadIdx.x & 31;
  if (gwave >= E) return;  // wave-uniform
  int s = src[gwave];
  int d = dst[gwave];
  int h = lane >> 3;
  float alpha = sc[(size_t)gwave * HEADS + h] / denom[(size_t)d * HEADS + h];
  float4 vv = ((const float4*)(vmat + (size_t)s * DIM))[lane];
  float* ap = agg + (size_t)d * DIM + lane * 4;
  atomicAdd(ap + 0, vv.x * alpha);
  atomicAdd(ap + 1, vv.y * alpha);
  atomicAdd(ap + 2, vv.z * alpha);
  atomicAdd(ap + 3, vv.w * alpha);
}

// ---------------------------------------------------------------------------
// Output projection: agg @ Wout + bout, ReLU, + x (residual).
// One wave per 16(M)x64(N) block; 2 column groups.
// ---------------------------------------------------------------------------
__global__ void out_wmma(const float* __restrict__ agg,
                         const float* __restrict__ Wout,
                         const float* __restrict__ bout,
                         const float* __restrict__ x,
                         float* __restrict__ out, int n) {
  int gwave = (blockIdx.x * blockDim.x + threadIdx.x) >> 5;
  int lane  = threadIdx.x & 31;
  int tileM = gwave >> 1;
  int g     = gwave & 1;
  int row0  = tileM * 16;
  if (row0 >= n) return;

  int colbase = g * 64;
  int lm = lane & 15;
  int lh = lane >> 4;
  int arow = row0 + lm;
  if (arow >= n) arow = n - 1;
  const float* grow = agg + (size_t)arow * DIM;

  v8f acc0 = {}, acc1 = {}, acc2 = {}, acc3 = {};
#pragma unroll
  for (int kk = 0; kk < DIM; kk += 32) {
    v16h a  = load_a_tile(grow, kk, lh);
    v16h b0 = load_b_tile(Wout, kk, lh, colbase + 0  + lm, DIM);
    v16h b1 = load_b_tile(Wout, kk, lh, colbase + 16 + lm, DIM);
    v16h b2 = load_b_tile(Wout, kk, lh, colbase + 32 + lm, DIM);
    v16h b3 = load_b_tile(Wout, kk, lh, colbase + 48 + lm, DIM);
    acc0 = __builtin_amdgcn_wmma_f32_16x16x32_f16(false, a, false, b0, (short)0, acc0, false, false);
    acc1 = __builtin_amdgcn_wmma_f32_16x16x32_f16(false, a, false, b1, (short)0, acc1, false, false);
    acc2 = __builtin_amdgcn_wmma_f32_16x16x32_f16(false, a, false, b2, (short)0, acc2, false, false);
    acc3 = __builtin_amdgcn_wmma_f32_16x16x32_f16(false, a, false, b3, (short)0, acc3, false, false);
  }

  float bias0 = bout[colbase + 0  + lm];
  float bias1 = bout[colbase + 16 + lm];
  float bias2 = bout[colbase + 32 + lm];
  float bias3 = bout[colbase + 48 + lm];
#pragma unroll
  for (int r = 0; r < 8; ++r) {
    int orow = row0 + lh * 8 + r;
    if (orow < n) {
      const float* xr = x   + (size_t)orow * DIM + colbase + lm;
      float*       orp = out + (size_t)orow * DIM + colbase + lm;
      orp[0]  = fmaxf(acc0[r] + bias0, 0.0f) + xr[0];
      orp[16] = fmaxf(acc1[r] + bias1, 0.0f) + xr[16];
      orp[32] = fmaxf(acc2[r] + bias2, 0.0f) + xr[32];
      orp[48] = fmaxf(acc3[r] + bias3, 0.0f) + xr[48];
    }
  }
}

// ---------------------------------------------------------------------------
extern "C" void kernel_launch(void* const* d_in, const int* in_sizes, int n_in,
                              void* d_out, int out_size, void* d_ws, size_t ws_size,
                              hipStream_t stream) {
  const float* x    = (const float*)d_in[0];
  const int*   ei   = (const int*)d_in[1];
  const float* Wt   = (const float*)d_in[2];
  const float* Wsm  = (const float*)d_in[3];
  const float* Wcm  = (const float*)d_in[4];
  const float* Wout = (const float*)d_in[5];
  const float* bout = (const float*)d_in[6];
  float*       out  = (float*)d_out;

  int n = in_sizes[0] / DIM;
  int E = in_sizes[1] / 2;
  const int* src = ei;
  const int* dst = ei + E;

  // Workspace layout (floats): q | k | v | agg | m | denom | scores
  float* ws    = (float*)d_ws;
  float* q     = ws;
  float* kmat  = q + (size_t)n * DIM;
  float* vmat  = kmat + (size_t)n * DIM;
  float* agg   = vmat + (size_t)n * DIM;
  float* m     = agg + (size_t)n * DIM;
  float* denom = m + (size_t)n * HEADS;
  float* sc    = denom + (size_t)n * HEADS;

  const int TB = 256;  // 8 waves/block (wave32)

  // init
  int init_blocks = (n * DIM + TB - 1) / TB;
  hipLaunchKernelGGL(init_ws, dim3(init_blocks), dim3(TB), 0, stream, m, denom, agg, n);

  // q/k/v projection: mtiles * 6 waves
  int mtiles = (n + 15) / 16;
  int qkv_waves = mtiles * 6;
  int qkv_blocks = (qkv_waves + 7) / 8;
  hipLaunchKernelGGL(qkv_wmma, dim3(qkv_blocks), dim3(TB), 0, stream,
                     x, Wt, Wsm, Wcm, q, kmat, vmat, n);

  // edge phase
  int ewave_blocks = (E + 7) / 8;       // wave-per-edge kernels
  int eh = E * HEADS;
  int eth_blocks = (eh + TB - 1) / TB;  // thread-per-(edge,head) kernel
  hipLaunchKernelGGL(edge_scores, dim3(ewave_blocks), dim3(TB), 0, stream,
                     q, kmat, src, dst, sc, m, E);
  hipLaunchKernelGGL(edge_exp, dim3(eth_blocks), dim3(TB), 0, stream,
                     dst, sc, m, denom, E);
  hipLaunchKernelGGL(edge_scatter, dim3(ewave_blocks), dim3(TB), 0, stream,
                     src, dst, sc, denom, vmat, agg, E);

  // output projection + epilogue: mtiles * 2 waves
  int out_waves = mtiles * 2;
  int out_blocks = (out_waves + 7) / 8;
  hipLaunchKernelGGL(out_wmma, dim3(out_blocks), dim3(TB), 0, stream,
                     agg, Wout, bout, x, out, n);
}
